// PatientTrajectoryDecoder_64330020159746
// MI455X (gfx1250) — matrix operand order repaired and checked by reading, not compiled, not run on hardware
//
#include <hip/hip_runtime.h>
#include <hip/hip_bf16.h>

// Problem constants (compile-time; reference fixes B,H,O,T)
#define BDIM 512
#define HDIM 1024
#define ODIM 256
#define TLEN 128
#define GDIM (4 * HDIM)   // 4096

typedef __attribute__((ext_vector_type(16))) __bf16 bf16x16;
typedef __attribute__((ext_vector_type(8)))  __bf16 bf16x8;
typedef __attribute__((ext_vector_type(8)))  float  f32x8;

// -------- workspace layout (bytes) --------
// bf16 weights + summed bias + recurrent state.  Total ~14.8 MB (L2-resident).
#define WS_WHH   ((size_t)0)                          // 4096*1024 bf16 = 8 MiB
#define WS_WIH   (WS_WHH  + (size_t)GDIM * HDIM * 2)  // 4096*256  bf16 = 2 MiB
#define WS_WOUT  (WS_WIH  + (size_t)GDIM * ODIM * 2)  // 256*1024  bf16 = 0.5 MiB
#define WS_BSUM  (WS_WOUT + (size_t)ODIM * HDIM * 2)  // 4096 f32
#define WS_H0    (WS_BSUM + (size_t)GDIM * 4)         // 512*1024 bf16 (h ping)
#define WS_H1    (WS_H0   + (size_t)BDIM * HDIM * 2)  // 512*1024 bf16 (h pong)
#define WS_C     (WS_H1   + (size_t)BDIM * HDIM * 2)  // 512*1024 f32  (cell)
#define WS_X     (WS_C    + (size_t)BDIM * HDIM * 4)  // 512*256  bf16 (fed-back pred)

__device__ __forceinline__ float sigf(float x) {
    return 1.0f / (1.0f + __expf(-x));
}

// A-fragment (16x32 bf16) per ISA layout: lane half selects K-halves
// {0..7,16..23} (lanes 0-15) or {8..15,24..31} (lanes 16-31).
// rowbase already points at (row m, element hi*8).
__device__ __forceinline__ bf16x16 load_a_frag(const __bf16* rowbase, int k0) {
    union { bf16x16 v; bf16x8 h[2]; } u;
    u.h[0] = *(const bf16x8*)(rowbase + k0);
    u.h[1] = *(const bf16x8*)(rowbase + k0 + 16);
    return u.v;
}

// ---------------------------------------------------------------------------
// Prep: fp32 weights -> bf16, b = b_ih + b_hh, h0 = bf16(enc), c = 0, x = 0
// ---------------------------------------------------------------------------
__global__ void lstm_prep_kernel(const float* __restrict__ Wih,
                                 const float* __restrict__ Whh,
                                 const float* __restrict__ bih,
                                 const float* __restrict__ bhh,
                                 const float* __restrict__ Wout,
                                 const float* __restrict__ enc,
                                 unsigned char* __restrict__ ws) {
    __bf16* whh  = (__bf16*)(ws + WS_WHH);
    __bf16* wih  = (__bf16*)(ws + WS_WIH);
    __bf16* wout = (__bf16*)(ws + WS_WOUT);
    float*  bsum = (float*)(ws + WS_BSUM);
    __bf16* h0   = (__bf16*)(ws + WS_H0);
    float*  cst  = (float*)(ws + WS_C);
    __bf16* xb   = (__bf16*)(ws + WS_X);

    size_t i = (size_t)blockIdx.x * blockDim.x + threadIdx.x;
    if (i < (size_t)GDIM * HDIM) whh[i]  = (__bf16)Whh[i];
    if (i < (size_t)GDIM * ODIM) wih[i]  = (__bf16)Wih[i];
    if (i < (size_t)ODIM * HDIM) wout[i] = (__bf16)Wout[i];
    if (i < (size_t)GDIM)        bsum[i] = bih[i] + bhh[i];
    if (i < (size_t)BDIM * HDIM) { h0[i] = (__bf16)enc[i]; cst[i] = 0.0f; }
    if (i < (size_t)BDIM * ODIM) xb[i] = (__bf16)0.0f;
}

// ---------------------------------------------------------------------------
// Gate kernel: one step of gates = x@W_ih^T + h@W_hh^T + b, fused LSTM cell.
// grid = (8 col-groups of 128 H-columns, 16 row-groups of 32 batch rows)
// block = 256 threads = 8 waves; wave w owns N-tile (colgrp*128 + w*16) for
// BOTH M=16 row tiles (weight fragments reused across M).
// No LDS, no barriers: A-frags are read straight from the WGP$-resident h/x.
// ---------------------------------------------------------------------------
__launch_bounds__(256, 1)
__global__ void lstm_gate_kernel(const __bf16* __restrict__ h_in,
                                 const __bf16* __restrict__ x_in,
                                 float* __restrict__ cst,
                                 const unsigned char* __restrict__ ws,
                                 __bf16* __restrict__ h_out) {
    const __bf16* whh  = (const __bf16*)(ws + WS_WHH);
    const __bf16* wih  = (const __bf16*)(ws + WS_WIH);
    const float*  bsum = (const float*)(ws + WS_BSUM);

    const int lane = threadIdx.x & 31;
    const int wave = threadIdx.x >> 5;
    const int ln   = lane & 15;
    const int hi   = lane >> 4;
    const int r0   = blockIdx.y * 32;                    // 32 batch rows
    const int ncol = blockIdx.x * 128 + wave * 16 + ln;  // column within H

    // A-row pointers for the two M-tiles (h and x inputs)
    const __bf16* a0h = h_in + (size_t)(r0 + ln) * HDIM + hi * 8;
    const __bf16* a1h = h_in + (size_t)(r0 + 16 + ln) * HDIM + hi * 8;
    const __bf16* a0x = x_in + (size_t)(r0 + ln) * ODIM + hi * 8;
    const __bf16* a1x = x_in + (size_t)(r0 + 16 + ln) * ODIM + hi * 8;

    // B-fragment row pointers: W row (gate*H + ncol), K halves as in A-fragment
    const __bf16* wp[4], * xp[4];
#pragma unroll
    for (int g = 0; g < 4; ++g) {
        wp[g] = whh + ((size_t)(g * HDIM + ncol)) * HDIM + hi * 8;
        xp[g] = wih + ((size_t)(g * HDIM + ncol)) * ODIM + hi * 8;
    }

    f32x8 acc[2][4] = {};

    // --- h @ W_hh^T : K over HDIM ---
#pragma unroll 2
    for (int k0 = 0; k0 < HDIM; k0 += 32) {
        if ((k0 & 63) == 0) {   // prefetch streamed weights ~1KB ahead
            __builtin_prefetch(wp[0] + k0 + 512, 0, 1);
            __builtin_prefetch(wp[1] + k0 + 512, 0, 1);
            __builtin_prefetch(wp[2] + k0 + 512, 0, 1);
            __builtin_prefetch(wp[3] + k0 + 512, 0, 1);
        }
        bf16x16 a0 = load_a_frag(a0h, k0);
        bf16x16 a1 = load_a_frag(a1h, k0);
#pragma unroll
        for (int g = 0; g < 4; ++g) {
            bf16x16 b = load_a_frag(wp[g], k0);
            acc[0][g] = __builtin_amdgcn_wmma_f32_16x16x32_bf16(false, a0, false, b, (short)0, acc[0][g], false, false);
            acc[1][g] = __builtin_amdgcn_wmma_f32_16x16x32_bf16(false, a1, false, b, (short)0, acc[1][g], false, false);
        }
    }
    // --- x @ W_ih^T : K over ODIM ---
#pragma unroll 2
    for (int k0 = 0; k0 < ODIM; k0 += 32) {
        bf16x16 a0 = load_a_frag(a0x, k0);
        bf16x16 a1 = load_a_frag(a1x, k0);
#pragma unroll
        for (int g = 0; g < 4; ++g) {
            bf16x16 b = load_a_frag(xp[g], k0);
            acc[0][g] = __builtin_amdgcn_wmma_f32_16x16x32_bf16(false, a0, false, b, (short)0, acc[0][g], false, false);
            acc[1][g] = __builtin_amdgcn_wmma_f32_16x16x32_bf16(false, a1, false, b, (short)0, acc[1][g], false, false);
        }
    }

    // --- fused bias + LSTM cell (gate order i,f,g,o); unique (row,col) owner ---
    const float bi_s = bsum[0 * HDIM + ncol];
    const float bf_s = bsum[1 * HDIM + ncol];
    const float bg_s = bsum[2 * HDIM + ncol];
    const float bo_s = bsum[3 * HDIM + ncol];
#pragma unroll
    for (int mt = 0; mt < 2; ++mt) {
#pragma unroll
        for (int v = 0; v < 8; ++v) {
            const int row = r0 + mt * 16 + v + hi * 8;
            const size_t idx = (size_t)row * HDIM + ncol;
            float gi = sigf(acc[mt][0][v] + bi_s);
            float gf = sigf(acc[mt][1][v] + bf_s);
            float gg = tanhf(acc[mt][2][v] + bg_s);
            float go = sigf(acc[mt][3][v] + bo_s);
            float cn = gf * cst[idx] + gi * gg;
            cst[idx]   = cn;
            h_out[idx] = (__bf16)(go * tanhf(cn));
        }
    }
}

// ---------------------------------------------------------------------------
// Pred kernel: pred = h@W_out^T + b_out; writes d_out[:, t, :] and feeds back
// pred as next x (bf16).  grid = (2 col-groups of 128 O-cols, 16 row-groups).
// ---------------------------------------------------------------------------
__launch_bounds__(256, 1)
__global__ void lstm_pred_kernel(const __bf16* __restrict__ h_in,
                                 const unsigned char* __restrict__ ws,
                                 const float* __restrict__ bout,
                                 float* __restrict__ out,
                                 __bf16* __restrict__ x_out,
                                 int t) {
    const __bf16* wout = (const __bf16*)(ws + WS_WOUT);

    const int lane = threadIdx.x & 31;
    const int wave = threadIdx.x >> 5;
    const int ln   = lane & 15;
    const int hi   = lane >> 4;
    const int r0   = blockIdx.y * 32;
    const int ncol = blockIdx.x * 128 + wave * 16 + ln;  // column within O

    const __bf16* a0h = h_in + (size_t)(r0 + ln) * HDIM + hi * 8;
    const __bf16* a1h = h_in + (size_t)(r0 + 16 + ln) * HDIM + hi * 8;
    const __bf16* wp  = wout + (size_t)ncol * HDIM + hi * 8;

    f32x8 acc[2] = {};
#pragma unroll 2
    for (int k0 = 0; k0 < HDIM; k0 += 32) {
        if ((k0 & 63) == 0) __builtin_prefetch(wp + k0 + 512, 0, 1);
        bf16x16 a0 = load_a_frag(a0h, k0);
        bf16x16 a1 = load_a_frag(a1h, k0);
        bf16x16 b  = load_a_frag(wp, k0);
        acc[0] = __builtin_amdgcn_wmma_f32_16x16x32_bf16(false, a0, false, b, (short)0, acc[0], false, false);
        acc[1] = __builtin_amdgcn_wmma_f32_16x16x32_bf16(false, a1, false, b, (short)0, acc[1], false, false);
    }

    const float bb = bout[ncol];
#pragma unroll
    for (int mt = 0; mt < 2; ++mt) {
#pragma unroll
        for (int v = 0; v < 8; ++v) {
            const int row = r0 + mt * 16 + v + hi * 8;
            const float p = acc[mt][v] + bb;
            out[((size_t)row * TLEN + t) * ODIM + ncol] = p;   // (B, T, O)
            x_out[(size_t)row * ODIM + ncol] = (__bf16)p;      // next-step x
        }
    }
}

// ---------------------------------------------------------------------------
// Inputs (setup_inputs order):
//  0 encoder_hidden (B,H) f32 | 1 target_length (int,1) | 2 W_ih (4H,O) f32
//  3 W_hh (4H,H) f32 | 4 b_ih (4H) | 5 b_hh (4H) | 6 W_out (O,H) | 7 b_out (O)
// Output: (B, T, O) f32
// ---------------------------------------------------------------------------
extern "C" void kernel_launch(void* const* d_in, const int* in_sizes, int n_in,
                              void* d_out, int out_size, void* d_ws, size_t ws_size,
                              hipStream_t stream) {
    const float* enc  = (const float*)d_in[0];
    const float* Wih  = (const float*)d_in[2];
    const float* Whh  = (const float*)d_in[3];
    const float* bih  = (const float*)d_in[4];
    const float* bhh  = (const float*)d_in[5];
    const float* Wout = (const float*)d_in[6];
    const float* bout = (const float*)d_in[7];
    float* out = (float*)d_out;
    unsigned char* ws = (unsigned char*)d_ws;

    __bf16* h0  = (__bf16*)(ws + WS_H0);
    __bf16* h1  = (__bf16*)(ws + WS_H1);
    float*  cst = (float*)(ws + WS_C);
    __bf16* xb  = (__bf16*)(ws + WS_X);

    // prep + state init (largest range: GDIM*HDIM = 4M elements)
    const int pt = 256;
    const int pb = (GDIM * HDIM + pt - 1) / pt;
    lstm_prep_kernel<<<pb, pt, 0, stream>>>(Wih, Whh, bih, bhh, Wout, enc, ws);

    // per-step kernels; kernel boundary = device-wide sync for the recurrence
    for (int t = 0; t < TLEN; ++t) {
        const __bf16* hin  = (t & 1) ? h1 : h0;
        __bf16*       hout = (t & 1) ? h0 : h1;
        lstm_gate_kernel<<<dim3(8, 16), 256, 0, stream>>>(hin, xb, cst, ws, hout);
        lstm_pred_kernel<<<dim3(2, 16), 256, 0, stream>>>(hout, ws, bout, out, xb, t);
    }
}
